// MaskBasedSingleDirectionRecurrentLayer_21698174779455
// MI455X (gfx1250) — matrix operand order repaired and checked by reading, not compiled, not run on hardware
//
#include <hip/hip_runtime.h>

// ---------------------------------------------------------------------------
// Masked single-direction LSTM for MI455X (gfx1250, wave32, WMMA)
//   gates = [x_t ; h_{t-1}] @ [W_ih ; W_hh]^T + (b_ih + b_hh)
// Persistent kernel: 32 WGs x 256 threads (8 wave32), weights in LDS,
// v_wmma_f32_16x16x32_f16 for all matrix math, one global barrier per step.
// ---------------------------------------------------------------------------

typedef _Float16 half8   __attribute__((ext_vector_type(8)));
typedef _Float16 half16  __attribute__((ext_vector_type(16)));
typedef float    floatx8 __attribute__((ext_vector_type(8)));

#define BB 32
#define TT 1024
#define II 512
#define HH 512
#define GG 2048          // 4*H
#define KTOT 1024        // I + H (fused K)
#define NWG 32           // persistent workgroups (one hidden-unit tile each)
#define NTHREADS 256     // 8 wave32 per WG

// LDS: weights 64 cols x 1024 K f16 (128KB) + gate staging 32x64 f32 (8KB)
//      + c state 512 f32 (2KB) + h state 512 f32 (2KB)
#define SMEM_W_HALFS   (64 * KTOT)
#define SMEM_BYTES     (SMEM_W_HALFS * 2 + 32 * 64 * 4 + 512 * 4 + 512 * 4)

// ---------------- workspace layout (bytes) ----------------
#define WS_BAR   ((size_t)0)                                   // 2 x u32 barrier
#define WS_WT    ((size_t)256)                                 // f16 [G][KTOT]  (4 MB)
#define WS_BSUM  (WS_WT + (size_t)GG * KTOT * 2)               // f32 [G]        (8 KB)
#define WS_XH    (WS_BSUM + (size_t)GG * 4)                    // f16 [T][B][I]  (32 MB)
#define WS_H16   (WS_XH + (size_t)TT * BB * II * 2)            // f16 [2][B][H]  (64 KB)

__global__ void init_barrier_kernel(unsigned* __restrict__ bar) {
  if (threadIdx.x < 2) bar[threadIdx.x] = 0u;
}

// Wt[g][k] = f16( k < I ? W_ih[g][k] : W_hh[g][k-I] );  bsum[g] = b_ih+b_hh
__global__ void prep_weights_kernel(const float* __restrict__ W_ih,
                                    const float* __restrict__ W_hh,
                                    const float* __restrict__ b_ih,
                                    const float* __restrict__ b_hh,
                                    _Float16* __restrict__ Wt,
                                    float* __restrict__ bsum) {
  int idx = blockIdx.x * blockDim.x + threadIdx.x;   // 0 .. G*KTOT-1
  int g = idx >> 10;
  int k = idx & (KTOT - 1);
  float v = (k < II) ? W_ih[g * II + k] : W_hh[g * HH + (k - II)];
  Wt[idx] = (_Float16)v;
  if (k == 0) bsum[g] = b_ih[g] + b_hh[g];
}

// xh[t][b][i] = f16( x[b][t][i] )   (transpose for per-step row access)
__global__ void prep_x_kernel(const float* __restrict__ x,
                              _Float16* __restrict__ xh) {
  int idx = blockIdx.x * blockDim.x + threadIdx.x;   // 0 .. B*T*I-1
  int i = idx & (II - 1);
  int t = (idx >> 9) & (TT - 1);
  int b = idx >> 19;                                 // T*I = 2^19
  xh[((size_t)t * BB + b) * II + i] = (_Float16)x[idx];
}

__device__ inline void global_barrier(unsigned* bar, unsigned& ls) {
  __syncthreads();
  if (threadIdx.x == 0) {
    ls ^= 1u;
    unsigned old = __hip_atomic_fetch_add(&bar[0], 1u, __ATOMIC_ACQ_REL,
                                          __HIP_MEMORY_SCOPE_AGENT);
    if (old == (unsigned)(NWG - 1)) {
      __hip_atomic_store(&bar[0], 0u, __ATOMIC_RELAXED, __HIP_MEMORY_SCOPE_AGENT);
      __hip_atomic_store(&bar[1], ls, __ATOMIC_RELEASE, __HIP_MEMORY_SCOPE_AGENT);
    } else {
      while (__hip_atomic_load(&bar[1], __ATOMIC_ACQUIRE,
                               __HIP_MEMORY_SCOPE_AGENT) != ls)
        __builtin_amdgcn_s_sleep(1);
    }
  }
  __syncthreads();
}

// Branchless fast transcendentals: v_exp_f32 + v_rcp_f32, no EXEC branching.
__device__ inline float sigmoid_fast(float v) {
  return __builtin_amdgcn_rcpf(1.0f + __expf(-v));
}
__device__ inline float tanh_fast(float v) {
  // tanh(x) = 2*sigmoid(2x) - 1
  return __builtin_amdgcn_fmed3f(
      2.0f * __builtin_amdgcn_rcpf(1.0f + __expf(-2.0f * v)) - 1.0f,
      -1.0f, 1.0f);
}

__global__ void __launch_bounds__(NTHREADS, 1)
lstm_persistent_kernel(const _Float16* __restrict__ Wt,     // [G][KTOT] f16
                       const float*    __restrict__ bsum,   // [G]
                       const _Float16* __restrict__ xh,     // [T][B][I] f16
                       const int*      __restrict__ lengths,// [B]
                       _Float16*       __restrict__ h16,    // [2][B][H] f16
                       float*          __restrict__ out,    // B*T*H | B*H | B*H
                       unsigned*       __restrict__ bar) {
  extern __shared__ char smem_raw[];
  _Float16* wlds = (_Float16*)smem_raw;                        // chunked weights
  float*    gbuf = (float*)(smem_raw + SMEM_W_HALFS * 2);      // [32][64] gates
  float*    cbuf = (float*)(smem_raw + SMEM_W_HALFS * 2 + 32 * 64 * 4); // [512]
  float*    hbuf = cbuf + 512;                                 // [512]

  const int wg   = blockIdx.x;        // hidden-unit tile: units [wg*16, wg*16+16)
  const int tid  = threadIdx.x;
  const int lane = tid & 31;
  const int wv   = tid >> 5;          // wave 0..7
  const int mt   = wv & 1;            // M tile (batch rows 0-15 / 16-31)
  const int gt   = wv >> 1;           // gate tile 0..3 (i,f,g,o)

  // ---- load this WG's weight slab into LDS (chunked, conflict-free) ----
  // local col lc in [0,64): gate = lc>>4, unit = wg*16 + (lc&15)
  // LDS index for (lc, k): ((k>>3)*64 + lc)*8 + (k&7)
  for (int it = tid; it < 64 * (KTOT / 8); it += NTHREADS) {
    int lcw = it / (KTOT / 8);
    int q   = it - lcw * (KTOT / 8);
    int n   = (lcw >> 4) * HH + wg * 16 + (lcw & 15);  // global gate row in Wt
    half8 v = *(const half8*)(Wt + (size_t)n * KTOT + (size_t)q * 8);
    *(half8*)(wlds + ((size_t)q * 64 + lcw) * 8) = v;
  }
  // ---- zero persistent state ----
  for (int u = tid; u < 512; u += NTHREADS) {
    cbuf[u] = 0.0f;
    hbuf[u] = 0.0f;
    int b = u >> 4, j = wg * 16 + (u & 15);
    h16[(size_t)b * HH + j] = (_Float16)0.0f;
    h16[(size_t)BB * HH + (size_t)b * HH + j] = (_Float16)0.0f;
  }
  __threadfence();
  unsigned ls = 0u;
  global_barrier(bar, ls);

  const int   cloc = lane & 15;
  const int   ncol = gt * HH + wg * 16 + cloc;      // global gate column
  const float bias = bsum[ncol];
  const int   kbA  = (lane < 16) ? 0 : 16;          // A-frag K sub-base
  const int   kbB  = (lane < 16) ? 0 : 16;          // B-frag K sub-base
  const int   arow = mt * 16 + (lane & 15);         // batch row for A frag
  const int   lc   = gt * 16 + cloc;                // local weight column
  const int   crow0 = mt * 16 + ((lane >> 4) << 3); // C-layout row base

  // per-thread elementwise unit assignment (fixed across steps)
  const int u0 = tid * 2;
  const int b0 = u0 >> 4, b1 = (u0 + 1) >> 4;
  const int len0 = lengths[b0];
  const int len1 = lengths[b1];

  for (int t = 0; t < TT; ++t) {
    // ------------------ GEMM phase: gates = [x_t;h] @ W^T + bias ----------
    floatx8 acc = {bias, bias, bias, bias, bias, bias, bias, bias};
    const _Float16* xrow = xh + ((size_t)t * BB + arow) * II;
    const _Float16* hrow = h16 + (size_t)(t & 1) * BB * HH + (size_t)arow * HH;

#pragma unroll 4
    for (int kt = 0; kt < KTOT / 32; ++kt) {
      int k0 = kt * 32 + kbA;
      const _Float16* aptr = (kt < (II / 32)) ? (xrow + k0) : (hrow + (k0 - II));
      union { half16 v; half8 h[2]; } afr, bfr;
      afr.h[0] = *(const half8*)(aptr);        // K kb .. kb+7
      afr.h[1] = *(const half8*)(aptr + 8);    // K kb+8 .. kb+15
      int q0 = (kt * 32 + kbB) >> 3;           // B frag: 16 contiguous K
      bfr.h[0] = *(const half8*)(wlds + ((size_t)q0 * 64 + lc) * 8);
      bfr.h[1] = *(const half8*)(wlds + ((size_t)(q0 + 1) * 64 + lc) * 8);
      acc = __builtin_amdgcn_wmma_f32_16x16x32_f16(
          false, afr.v, false, bfr.v, (short)0, acc, false, false);
    }

    // store C tile to LDS gate buffer: rows = batch, cols = 64 local gates
#pragma unroll
    for (int r = 0; r < 8; ++r) {
      gbuf[(crow0 + r) * 64 + lc] = acc[r];
    }
    __syncthreads();

    // ------------------ elementwise LSTM update (WG-local) ----------------
#pragma unroll
    for (int rep = 0; rep < 2; ++rep) {
      int u  = u0 + rep;                 // 0..511
      int b  = (rep == 0) ? b0 : b1;
      int jc = u & 15;
      float ig = gbuf[b * 64 +      jc];
      float fg = gbuf[b * 64 + 16 + jc];
      float gg = gbuf[b * 64 + 32 + jc];
      float og = gbuf[b * 64 + 48 + jc];
      float c_old = cbuf[u];
      float h_old = hbuf[u];
      float c_new = sigmoid_fast(fg) * c_old + sigmoid_fast(ig) * tanh_fast(gg);
      float h_new = sigmoid_fast(og) * tanh_fast(c_new);
      bool  m  = (t < ((rep == 0) ? len0 : len1));
      float hb = m ? h_new : h_old;
      float cb = m ? c_new : c_old;
      cbuf[u] = cb;
      hbuf[u] = hb;
      int j = wg * 16 + jc;
      h16[(size_t)((t + 1) & 1) * BB * HH + (size_t)b * HH + j] = (_Float16)hb;
      out[((size_t)b * TT + t) * HH + j] = m ? h_new : 0.0f;
    }
    __threadfence();
    global_barrier(bar, ls);   // h16[(t+1)&1] visible to all WGs' next GEMM
  }

  // ------------------ final states h_n, c_n ------------------
  const size_t off_h = (size_t)BB * TT * HH;
  const size_t off_c = off_h + (size_t)BB * HH;
  for (int u = tid; u < 512; u += NTHREADS) {
    int b = u >> 4, j = wg * 16 + (u & 15);
    out[off_h + (size_t)b * HH + j] = hbuf[u];
    out[off_c + (size_t)b * HH + j] = cbuf[u];
  }
}

// ---------------------------------------------------------------------------
extern "C" void kernel_launch(void* const* d_in, const int* in_sizes, int n_in,
                              void* d_out, int out_size, void* d_ws, size_t ws_size,
                              hipStream_t stream) {
  const float* x       = (const float*)d_in[0];
  const int*   lengths = (const int*)  d_in[1];
  const float* W_ih    = (const float*)d_in[2];
  const float* W_hh    = (const float*)d_in[3];
  const float* b_ih    = (const float*)d_in[4];
  const float* b_hh    = (const float*)d_in[5];
  float*       out     = (float*)d_out;

  char*      ws    = (char*)d_ws;
  unsigned*  bar   = (unsigned*)(ws + WS_BAR);
  _Float16*  Wt    = (_Float16*)(ws + WS_WT);
  float*     bsum  = (float*)(ws + WS_BSUM);
  _Float16*  xh    = (_Float16*)(ws + WS_XH);
  _Float16*  h16   = (_Float16*)(ws + WS_H16);

  (void)in_sizes; (void)n_in; (void)out_size; (void)ws_size;

  // allow 140KB dynamic LDS (host-side, idempotent, not stream-captured)
  (void)hipFuncSetAttribute((const void*)lstm_persistent_kernel,
                            hipFuncAttributeMaxDynamicSharedMemorySize,
                            SMEM_BYTES);

  init_barrier_kernel<<<1, 64, 0, stream>>>(bar);
  prep_weights_kernel<<<(GG * KTOT) / 256, 256, 0, stream>>>(
      W_ih, W_hh, b_ih, b_hh, Wt, bsum);
  prep_x_kernel<<<(BB * TT * II) / 256, 256, 0, stream>>>(x, xh);
  lstm_persistent_kernel<<<NWG, NTHREADS, SMEM_BYTES, stream>>>(
      Wt, bsum, xh, lengths, h16, out, bar);
}
